// SWMSA_66297115181258
// MI455X (gfx1250) — matrix operand order repaired and checked by reading, not compiled, not run on hardware
//
#include <hip/hip_runtime.h>
#include <hip/hip_bf16.h>
#include <math.h>

// ---------------- problem constants ----------------
#define BN    4
#define HN    224
#define WN    224
#define CN    256
#define WINN  8
#define HEADS 8
#define HD    32
#define SHIFT 4
#define NHW   28          // H/WIN
#define NWIN  784         // NHW*NHW

// ---------------- workspace layout (bytes) ----------------
// wqbT: bf16 w_qkv transposed  [768][256]  (row = out col n, contiguous K)
// wpbT: bf16 w_proj transposed [256][256]
#define WQ_OFF   0ull
#define WP_OFF   (WQ_OFF + 768ull*256*2)
#define BT_OFF   (WP_OFF + 256ull*256*2)
#define QKV_OFF  (BT_OFF + 8192ull)                            // btab f32 225*8
#define XO_OFF   (QKV_OFF + (size_t)BN*NWIN*HEADS*3*64*HD*4)   // qkv f32
// XO: f32 (B,H,W,C) un-rolled attention output

// ---------------- types & helpers ----------------
typedef __attribute__((ext_vector_type(16))) __bf16       bf16x16;
typedef __attribute__((ext_vector_type(8)))  __bf16       bf16x8;
typedef __attribute__((ext_vector_type(2)))  __bf16       bf16x2;
typedef __attribute__((ext_vector_type(8)))  float        f32x8;
typedef __attribute__((ext_vector_type(2)))  float        f32x2;
typedef __attribute__((ext_vector_type(8)))  unsigned int u32x8;

__device__ inline unsigned short f2bf(float f) {
  __bf16 h = (__bf16)f;
  return __builtin_bit_cast(unsigned short, h);
}

// packed 2x f32 -> 2x bf16 (lowers to v_cvt_pk_bf16_f32)
__device__ inline unsigned int pkbf(float lo, float hi) {
  f32x2 f;
  f[0] = lo; f[1] = hi;
  bf16x2 h = __builtin_convertvector(f, bf16x2);
  return __builtin_bit_cast(unsigned int, h);
}

// packed 8x f32 -> 8x bf16 (4x v_cvt_pk_bf16_f32)
__device__ inline uint4 cvt8(float4 a, float4 b) {
  f32x8 f;
  f[0] = a.x; f[1] = a.y; f[2] = a.z; f[3] = a.w;
  f[4] = b.x; f[5] = b.y; f[6] = b.z; f[7] = b.w;
  bf16x8 h = __builtin_convertvector(f, bf16x8);
  return __builtin_bit_cast(uint4, h);
}

__device__ inline f32x8 zero8() {
  f32x8 z;
  for (int i = 0; i < 8; ++i) z[i] = 0.0f;
  return z;
}

__device__ inline bf16x16 make_frag(const unsigned short* p0, const unsigned short* p1) {
  uint4 a = *(const uint4*)p0;
  uint4 b = *(const uint4*)p1;
  u32x8 u;
  u[0] = a.x; u[1] = a.y; u[2] = a.z; u[3] = a.w;
  u[4] = b.x; u[5] = b.y; u[6] = b.z; u[7] = b.w;
  return __builtin_bit_cast(bf16x16, u);
}

// A-matrix 16x32 bf16 fragment, source tile row-major [rows][kStride] ushort.
// ISA: lanes 0-15 M=lane, 16-31 M=lane-16; elems 0..7 -> K=hi*8+0..7, 8..15 -> K=16+hi*8+0..7
__device__ inline bf16x16 frag_a(const unsigned short* base, int row0, int kOff,
                                 int kStride, int lane) {
  int hi = (lane >> 4) & 1;
  int m  = lane & 15;
  const unsigned short* p = base + (size_t)(row0 + m) * kStride + kOff + hi * 8;
  return make_frag(p, p + 16);
}

// B-matrix 32x16 bf16 fragment from an [N][K]-transposed tile.
// ISA: lanes 0-15 N=lane, 16-31 N=lane-16; elems 0..7 -> K=hi*8+0..7, 8..15 -> K=16+hi*8+0..7
__device__ inline bf16x16 frag_b(const unsigned short* base, int col0, int kOff,
                                 int kStride, int lane) {
  int hi = (lane >> 4) & 1;
  int n  = lane & 15;
  const unsigned short* p = base + (size_t)(col0 + n) * kStride + kOff + hi * 8;
  return make_frag(p, p + 16);
}

__device__ inline f32x8 wmma_bf16(bf16x16 a, bf16x16 b, f32x8 c) {
  return __builtin_amdgcn_wmma_f32_16x16x32_bf16(false, a, false, b, (short)0, c,
                                                 false, false);
}

// ---------------- Tensor Data Mover (guarded) ----------------
#if __has_builtin(__builtin_amdgcn_tensor_load_to_lds) && \
    __has_builtin(__builtin_amdgcn_s_wait_tensorcnt)
#define HAVE_TDM 1
#else
#define HAVE_TDM 0
#endif

#if HAVE_TDM
typedef __attribute__((ext_vector_type(4))) unsigned int u32x4;
typedef __attribute__((ext_vector_type(8))) int          i32x8;
typedef __attribute__((ext_vector_type(4))) int          i32x4;

__device__ inline unsigned int lds_off(const void* p) {
  return (unsigned int)(unsigned long long)p;   // low 32 bits = LDS byte address
}

// 2D bf16 tile load: tensor [d1][d0] (d0 contiguous, row stride = d0),
// tile [tile_d1][tile_d0] starting at gsrc, written contiguously to LDS.
// D# bit layout per CDNA5 ISA ch.8.3/8.4.
__device__ inline void tdm_load_2d_bf16(const unsigned short* gsrc,
                                        unsigned int ldsOff,
                                        unsigned int d0, unsigned int d1,
                                        unsigned int tile_d0, unsigned int tile_d1) {
  unsigned long long ga = (unsigned long long)gsrc;
  u32x4 g0;
  g0[0] = 1u;                                            // count=1, no gather
  g0[1] = ldsOff;                                        // lds_addr
  g0[2] = (unsigned int)(ga & 0xffffffffull);            // global_addr[31:0]
  g0[3] = (unsigned int)((ga >> 32) & 0x1ffffffull)      // global_addr[56:32]
          | 0x80000000u;                                 // type=2 ("image")
  i32x8 g1;
  g1[0] = (int)0x00010000u;                              // data_size=1 (2 bytes)
  g1[1] = (int)(d0 << 16);                               // tensor_dim0[15:0]
  g1[2] = (int)((d0 >> 16) | (d1 << 16));                // d0[31:16] | d1[15:0]
  g1[3] = (int)((d1 >> 16) | (tile_d0 << 16));           // d1[31:16] | tile_dim0
  g1[4] = (int)tile_d1;                                  // tile_dim1 (tile_dim2=0)
  g1[5] = (int)d0;                                       // tensor_dim0_stride[31:0]
  g1[6] = 0;                                             // stride hi / dim1_stride lo
  g1[7] = 0;
  i32x4 g2; g2[0] = 0; g2[1] = 0; g2[2] = 0; g2[3] = 0;
  i32x4 g3; g3[0] = 0; g3[1] = 0; g3[2] = 0; g3[3] = 0;
#if __clang_major__ >= 23
  i32x8 gx;
  for (int i = 0; i < 8; ++i) gx[i] = 0;
  __builtin_amdgcn_tensor_load_to_lds(g0, g1, g2, g3, gx, 0);
#else
  __builtin_amdgcn_tensor_load_to_lds(g0, g1, g2, g3, 0);
#endif
}
#endif  // HAVE_TDM

// ---------------- kernel 0: weight fp32 -> bf16, transposed ----------------
__global__ void k_prep_weights(const float* __restrict__ wq, const float* __restrict__ wp,
                               unsigned short* __restrict__ wqbT,
                               unsigned short* __restrict__ wpbT) {
  int i = blockIdx.x * blockDim.x + threadIdx.x;
  int stride = gridDim.x * blockDim.x;
  for (int t = i; t < 768 * 256; t += stride) {
    int n = t >> 8, k = t & 255;
    wqbT[t] = f2bf(wq[(size_t)k * 768 + n]);
  }
  for (int t = i; t < 256 * 256; t += stride) {
    int n = t >> 8, k = t & 255;
    wpbT[t] = f2bf(wp[(size_t)k * 256 + n]);
  }
}

// ---------------- kernel 1: CPB MLP -> btab[225][8] ----------------
__device__ inline float relcoord(int idx) {
  float v = (float)(idx - 7) * (8.0f / 7.0f);
  float s = (v > 0.0f) ? 1.0f : ((v < 0.0f) ? -1.0f : 0.0f);
  return s * log1pf(fabsf(v)) * (1.0f / 2.0794415416798357f); // /log(8)
}

__global__ __launch_bounds__(64) void k_cpb(const float* __restrict__ w1,
                                            const float* __restrict__ b1,
                                            const float* __restrict__ w2,
                                            float* __restrict__ btab) {
  __shared__ float red[64][8];
  int t   = blockIdx.x;   // 0..224
  int tid = threadIdx.x;  // 0..63
  float r0 = relcoord(t / 15);
  float r1 = relcoord(t % 15);
  float acc[8];
  for (int e = 0; e < 8; ++e) acc[e] = 0.0f;
  for (int hh = tid * 8; hh < tid * 8 + 8; ++hh) {
    float h = fmaxf(0.0f, r0 * w1[hh] + r1 * w1[512 + hh] + b1[hh]);
    for (int e = 0; e < 8; ++e) acc[e] += h * w2[hh * 8 + e];
  }
  for (int e = 0; e < 8; ++e) red[tid][e] = acc[e];
  __syncthreads();
  if (tid < 8) {
    float s = 0.0f;
    for (int j = 0; j < 64; ++j) s += red[j][tid];
    btab[t * 8 + tid] = 16.0f / (1.0f + expf(-s));
  }
}

// ---------------- kernel 2: QKV GEMM fused with roll + window partition ------
// grid (6, B*NWIN), block 256 (8 waves). Block: 64 window-tokens x 128 cols, K=256.
__global__ __launch_bounds__(256) void k_qkv(const float* __restrict__ x,
                                             const unsigned short* __restrict__ wqbT,
                                             const float* __restrict__ qb,
                                             const float* __restrict__ vb,
                                             float* __restrict__ qkvws) {
  __shared__ __align__(16) unsigned short aT[64 * 32];
  __shared__ __align__(16) unsigned short bT[128 * 32];
  int bw  = blockIdx.y;
  int n0  = blockIdx.x * 128;
  int b   = bw / NWIN;
  int win = bw % NWIN;
  int wh  = win / NHW, ww = win % NHW;
  int tid = threadIdx.x;

  // A-load: this thread owns row ar, 8 contiguous K per step
  int ar = tid >> 2;
  int ac = (tid & 3) * 8;
  int hr = wh * 8 + (ar >> 3), wr = ww * 8 + (ar & 7);
  int h = (hr + SHIFT) % HN, w = (wr + SHIFT) % WN;
  const float* xrow = x + (((size_t)b * HN + h) * WN + w) * CN;

  int wave = tid >> 5, lane = tid & 31;
  int wr_ = wave & 3, wc = wave >> 2;

#if !HAVE_TDM
  int bn = tid >> 1;          // 0..127 (tile row = out col)
  int bo = (tid & 1) * 16;    // element offset in K
  const unsigned short* bsrcRow = wqbT + (size_t)(n0 + bn) * 256 + bo;
  uint4 bu0 = *(const uint4*)(bsrcRow);
  uint4 bu1 = *(const uint4*)(bsrcRow + 8);
#endif
  float4 fa0 = *(const float4*)(xrow + ac);
  float4 fa1 = *(const float4*)(xrow + ac + 4);

  f32x8 c[4];
  for (int t = 0; t < 4; ++t) c[t] = zero8();

  for (int k0 = 0; k0 < 256; k0 += 32) {
#if HAVE_TDM
    if (wave == 0)
      tdm_load_2d_bf16(wqbT + (size_t)n0 * 256 + k0, lds_off(bT), 256, 768, 32, 128);
#endif
    *(uint4*)&aT[ar * 32 + ac] = cvt8(fa0, fa1);
#if !HAVE_TDM
    *(uint4*)&bT[bn * 32 + bo]     = bu0;
    *(uint4*)&bT[bn * 32 + bo + 8] = bu1;
#endif
#if HAVE_TDM
    if (wave == 0) __builtin_amdgcn_s_wait_tensorcnt(0);
#endif
    __syncthreads();

    if (k0 + 32 < 256) {  // software pipeline: issue next tile's global loads now
      fa0 = *(const float4*)(xrow + k0 + 32 + ac);
      fa1 = *(const float4*)(xrow + k0 + 32 + ac + 4);
#if !HAVE_TDM
      bu0 = *(const uint4*)(bsrcRow + k0 + 32);
      bu1 = *(const uint4*)(bsrcRow + k0 + 32 + 8);
#endif
    }
    bf16x16 a = frag_a(aT, 16 * wr_, 0, 32, lane);
    for (int t = 0; t < 4; ++t)
      c[t] = wmma_bf16(a, frag_b(bT, 64 * wc + 16 * t, 0, 32, lane), c[t]);
    __syncthreads();
  }

  // epilogue: scatter into (bw, head, part, 64, 32) f32, add q/v bias
  int hi = lane >> 4, ln = lane & 15;
  for (int t = 0; t < 4; ++t) {
    int col  = n0 + 64 * wc + 16 * t + ln;
    int part = col >> 8;
    int dc   = col & 255;
    int head = dc >> 5, hd = dc & 31;
    float bias = (part == 0) ? qb[dc] : ((part == 2) ? vb[dc] : 0.0f);
    float* dst = qkvws + (((size_t)bw * HEADS + head) * 3 + part) * 2048 + hd;
    for (int r = 0; r < 8; ++r) {
      int m = 16 * wr_ + r + hi * 8;
      dst[(size_t)m * 32] = c[t][r] + bias;
    }
  }
}

// ---------------- kernel 3: windowed cosine attention ----------------
// grid (HEADS, B*NWIN), block 128 (4 waves). One (window, head) per block.
__global__ __launch_bounds__(128) void k_attn(const float* __restrict__ qkvws,
                                              const float* __restrict__ logit_scale,
                                              const float* __restrict__ btab,
                                              float* __restrict__ xo) {
  __shared__ __align__(16) unsigned short qn[64 * 32];
  __shared__ __align__(16) unsigned short kn[64 * 32];
  __shared__ __align__(16) unsigned short vT[32 * 64];
  __shared__ __align__(16) unsigned short pP[64 * 64];
  __shared__ int    regS[64];
  __shared__ size_t rowBase[64];
  __shared__ float  sExp;

  int head = blockIdx.x;
  int bw   = blockIdx.y;
  int b    = bw / NWIN;
  int win  = bw % NWIN;
  int wh   = win / NHW, ww = win % NHW;
  int tid  = threadIdx.x;
  const float* base = qkvws + ((size_t)bw * HEADS + head) * 3 * 2048;

  if (tid == 0) sExp = expf(logit_scale[head]);

  if (tid < 64) {
    int m = tid;
    const float* qr = base + m * 32;
    float ss = 0.0f;
    for (int d = 0; d < 32; ++d) ss += qr[d] * qr[d];
    float inv = rsqrtf(fmaxf(ss, 1.55e-05f));
    for (int d = 0; d < 32; d += 2)
      *(unsigned int*)&qn[m * 32 + d] = pkbf(qr[d] * inv, qr[d + 1] * inv);

    int hr = wh * 8 + (m >> 3), wr = ww * 8 + (m & 7);
    int rrow = (hr < HN - WINN) ? 0 : ((hr < HN - SHIFT) ? 1 : 2);
    int rcol = (wr < WN - WINN) ? 0 : ((wr < WN - SHIFT) ? 1 : 2);
    regS[m] = rrow * 3 + rcol;
    int h = (hr + SHIFT) % HN, w2 = (wr + SHIFT) % WN;
    rowBase[m] = (((size_t)b * HN + h) * WN + w2) * CN;
  } else {
    int m = tid - 64;
    const float* kr = base + 2048 + m * 32;
    float ss = 0.0f;
    for (int d = 0; d < 32; ++d) ss += kr[d] * kr[d];
    float inv = rsqrtf(fmaxf(ss, 1.55e-05f));
    for (int d = 0; d < 32; d += 2)
      *(unsigned int*)&kn[m * 32 + d] = pkbf(kr[d] * inv, kr[d + 1] * inv);
  }
  {
    const float* vr = base + 2 * 2048;
    int hd = tid >> 2;
    int k0 = (tid & 3) * 16;
    for (int i = 0; i < 16; i += 2) {
      int ki = k0 + i;
      *(unsigned int*)&vT[hd * 64 + ki] = pkbf(vr[ki * 32 + hd], vr[(ki + 1) * 32 + hd]);
    }
  }
  __syncthreads();

  int wv = tid >> 5, lane = tid & 31, hi = lane >> 4, ln = lane & 15;

  // S = qn @ kn^T : single K=32 step, 4 N-tiles
  bf16x16 qa = frag_a(qn, 16 * wv, 0, 32, lane);
  f32x8 s[4];
  for (int t = 0; t < 4; ++t)
    s[t] = wmma_bf16(qa, frag_b(kn, 16 * t, 0, 32, lane), zero8());

  float eS = sExp;
  for (int t = 0; t < 4; ++t)
    for (int r = 0; r < 8; ++r) {
      int qi = 16 * wv + r + hi * 8;
      int ki = 16 * t + ln;
      int rel = ((qi >> 3) - (ki >> 3) + 7) * 15 + ((qi & 7) - (ki & 7) + 7);
      float mask = (regS[qi] == regS[ki]) ? 0.0f : -100.0f;
      s[t][r] = s[t][r] * eS + btab[rel * 8 + head] + mask;
    }

  // row softmax (row lives in one VGPR index across 16 lanes of a half-wave)
  for (int r = 0; r < 8; ++r) {
    float mx = -3.0e38f;
    for (int t = 0; t < 4; ++t) mx = fmaxf(mx, s[t][r]);
    for (int off = 8; off >= 1; off >>= 1) mx = fmaxf(mx, __shfl_xor(mx, off, 32));
    float sum = 0.0f;
    for (int t = 0; t < 4; ++t) {
      float e = __expf(s[t][r] - mx);
      s[t][r] = e;
      sum += e;
    }
    for (int off = 8; off >= 1; off >>= 1) sum += __shfl_xor(sum, off, 32);
    float invs = 1.0f / sum;
    for (int t = 0; t < 4; ++t) s[t][r] *= invs;
  }

  // P -> LDS bf16
  for (int t = 0; t < 4; ++t)
    for (int r = 0; r < 8; r += 2)
      ; // (kept simple below)
  for (int t = 0; t < 4; ++t)
    for (int r = 0; r < 8; ++r)
      pP[(16 * wv + r + hi * 8) * 64 + 16 * t + ln] = f2bf(s[t][r]);
  __syncthreads();

  // O = P @ V : K=64 in two steps, 2 N-tiles of HD
  f32x8 o[2];
  o[0] = zero8(); o[1] = zero8();
  for (int kk = 0; kk < 64; kk += 32) {
    bf16x16 pa = frag_a(pP, 16 * wv, kk, 64, lane);
    for (int nt = 0; nt < 2; ++nt)
      o[nt] = wmma_bf16(pa, frag_b(vT, 16 * nt, kk, 64, lane), o[nt]);
  }

  // write back into un-rolled (B,H,W,C) layout
  for (int nt = 0; nt < 2; ++nt)
    for (int r = 0; r < 8; ++r) {
      int qi = 16 * wv + r + hi * 8;
      int hd = 16 * nt + ln;
      xo[rowBase[qi] + head * 32 + hd] = o[nt][r];
    }
}

// ---------------- kernel 4: output projection GEMM + bias ----------------
// grid (2, NTOK/64), block 256 (8 waves). 64x128 tile, K=256.
__global__ __launch_bounds__(256) void k_proj(const float* __restrict__ xo,
                                              const unsigned short* __restrict__ wpbT,
                                              const float* __restrict__ bp,
                                              float* __restrict__ out) {
  __shared__ __align__(16) unsigned short aT[64 * 32];
  __shared__ __align__(16) unsigned short bT[128 * 32];
  int m0  = blockIdx.y * 64;
  int n0  = blockIdx.x * 128;
  int tid = threadIdx.x;

  int ar = tid >> 2;
  int ac = (tid & 3) * 8;
  const float* xrow = xo + (size_t)(m0 + ar) * 256;
  int wave = tid >> 5, lane = tid & 31;
  int wr_ = wave & 3, wc = wave >> 2;

#if !HAVE_TDM
  int bn = tid >> 1;
  int bo = (tid & 1) * 16;
  const unsigned short* bsrcRow = wpbT + (size_t)(n0 + bn) * 256 + bo;
  uint4 bu0 = *(const uint4*)(bsrcRow);
  uint4 bu1 = *(const uint4*)(bsrcRow + 8);
#endif
  float4 fa0 = *(const float4*)(xrow + ac);
  float4 fa1 = *(const float4*)(xrow + ac + 4);

  f32x8 c[4];
  for (int t = 0; t < 4; ++t) c[t] = zero8();

  for (int k0 = 0; k0 < 256; k0 += 32) {
#if HAVE_TDM
    if (wave == 0)
      tdm_load_2d_bf16(wpbT + (size_t)n0 * 256 + k0, lds_off(bT), 256, 256, 32, 128);
#endif
    *(uint4*)&aT[ar * 32 + ac] = cvt8(fa0, fa1);
#if !HAVE_TDM
    *(uint4*)&bT[bn * 32 + bo]     = bu0;
    *(uint4*)&bT[bn * 32 + bo + 8] = bu1;
#endif
#if HAVE_TDM
    if (wave == 0) __builtin_amdgcn_s_wait_tensorcnt(0);
#endif
    __syncthreads();

    if (k0 + 32 < 256) {
      fa0 = *(const float4*)(xrow + k0 + 32 + ac);
      fa1 = *(const float4*)(xrow + k0 + 32 + ac + 4);
#if !HAVE_TDM
      bu0 = *(const uint4*)(bsrcRow + k0 + 32);
      bu1 = *(const uint4*)(bsrcRow + k0 + 32 + 8);
#endif
    }
    bf16x16 a = frag_a(aT, 16 * wr_, 0, 32, lane);
    for (int t = 0; t < 4; ++t)
      c[t] = wmma_bf16(a, frag_b(bT, 64 * wc + 16 * t, 0, 32, lane), c[t]);
    __syncthreads();
  }

  int hi = lane >> 4, ln = lane & 15;
  for (int t = 0; t < 4; ++t) {
    int col = n0 + 64 * wc + 16 * t + ln;
    float bias = bp[col];
    for (int r = 0; r < 8; ++r) {
      int row = m0 + 16 * wr_ + r + hi * 8;
      out[(size_t)row * 256 + col] = c[t][r] + bias;
    }
  }
}

// ---------------- launcher ----------------
extern "C" void kernel_launch(void* const* d_in, const int* in_sizes, int n_in,
                              void* d_out, int out_size, void* d_ws, size_t ws_size,
                              hipStream_t stream) {
  const float* x      = (const float*)d_in[0];
  const float* w_qkv  = (const float*)d_in[1];
  const float* q_bias = (const float*)d_in[2];
  const float* v_bias = (const float*)d_in[3];
  const float* scale  = (const float*)d_in[4];
  const float* cpb_w1 = (const float*)d_in[5];
  const float* cpb_b1 = (const float*)d_in[6];
  const float* cpb_w2 = (const float*)d_in[7];
  const float* w_proj = (const float*)d_in[8];
  const float* b_proj = (const float*)d_in[9];

  char* ws = (char*)d_ws;
  unsigned short* wqbT  = (unsigned short*)(ws + WQ_OFF);
  unsigned short* wpbT  = (unsigned short*)(ws + WP_OFF);
  float*          btab  = (float*)(ws + BT_OFF);
  float*          qkvws = (float*)(ws + QKV_OFF);
  float*          xo    = (float*)(ws + XO_OFF);

  k_prep_weights<<<768, 256, 0, stream>>>(w_qkv, w_proj, wqbT, wpbT);
  k_cpb<<<225, 64, 0, stream>>>(cpb_w1, cpb_b1, cpb_w2, btab);
  k_qkv<<<dim3(6, BN * NWIN), 256, 0, stream>>>(x, wqbT, q_bias, v_bias, qkvws);
  k_attn<<<dim3(HEADS, BN * NWIN), 128, 0, stream>>>(qkvws, scale, btab, xo);
  k_proj<<<dim3(2, (BN * HN * WN) / 64), 256, 0, stream>>>(xo, wpbT, b_proj,
                                                           (float*)d_out);
}
